// Attention_71073118814268
// MI455X (gfx1250) — matrix operand order repaired and checked by reading, not compiled, not run on hardware
//
#include <hip/hip_runtime.h>
#include <hip/hip_bf16.h>

// ---------------------------------------------------------------------------
// Shapes (fixed by the reference):
//   B=2, N=2048, D=1024, H=16, Dh=64, inner=H*Dh=1024, Mrows=B*N=4096
// ---------------------------------------------------------------------------
#define BB    2
#define NN    2048
#define DD    1024
#define HH    16
#define DHD   64
#define INNER 1024
#define MROWS 4096

typedef unsigned int       u32;
typedef unsigned long long u64;
typedef __attribute__((ext_vector_type(8)))  __bf16 v8bf;
typedef __attribute__((ext_vector_type(16))) __bf16 v16bf;
typedef __attribute__((ext_vector_type(8)))  float  v8f;
typedef __attribute__((ext_vector_type(4)))  unsigned int u32x4;
typedef __attribute__((ext_vector_type(8)))  int    i32x8;
typedef __attribute__((ext_vector_type(4)))  int    i32x4;

#ifndef __has_builtin
#define __has_builtin(x) 0
#endif

#if __has_builtin(__builtin_amdgcn_tensor_load_to_lds)
#define HAVE_TDM 1
#else
#define HAVE_TDM 0
#endif

// therock-10.0 headers ship the 6-arg TDM builtin; ROCm 7.2 has the 5-arg one.
#if __has_include(<hip/amd_detail/amd_gfx1250_TDM.h>)
#define TDM_SIX_ARGS 1
#else
#define TDM_SIX_ARGS 0
#endif

#if __has_builtin(__builtin_amdgcn_s_wait_tensorcnt)
#define WAIT_TENSORCNT(n) __builtin_amdgcn_s_wait_tensorcnt(n)
#else
#define WAIT_TENSORCNT(n) asm volatile("s_wait_tensorcnt " #n ::: "memory")
#endif

#if __has_builtin(__builtin_amdgcn_permlane16)
#define HAVE_PERMLANE16 1
#else
#define HAVE_PERMLANE16 0
#endif

// ---------------------------------------------------------------------------
// Small helpers
// ---------------------------------------------------------------------------
static __device__ __forceinline__ v8f v8f_zero() {
    v8f z;
#pragma unroll
    for (int i = 0; i < 8; ++i) z[i] = 0.0f;
    return z;
}

static __device__ __forceinline__ v16bf combine16(v8bf lo, v8bf hi) {
    union { v16bf v; v8bf h[2]; } u;
    u.h[0] = lo; u.h[1] = hi;
    return u.v;
}

// A-matrix (16x32 bf16) fragment: lane L: M = L%16, half = L/16.
// e0..7  -> K = 8*half + e ; e8..15 -> K = 16 + 8*half + (e-8)
template <typename P>
static __device__ __forceinline__ v16bf frag_a(const P* rowbase, int half) {
    v8bf lo = *(const v8bf*)(rowbase + 8 * half);
    v8bf hi = *(const v8bf*)(rowbase + 16 + 8 * half);
    return combine16(lo, hi);
}

// B-matrix (32x16 bf16) fragment: lane L: N = L%16, half = L/16.
// e0..7  -> K = 8*half + e ; e8..15 -> K = 16 + 8*half + (e-8). colbase = K-major vector of column N.
template <typename P>
static __device__ __forceinline__ v16bf frag_b(const P* colbase, int half) {
    v8bf lo = *(const v8bf*)(colbase + 8 * half);
    v8bf hi = *(const v8bf*)(colbase + 16 + 8 * half);
    return combine16(lo, hi);
}

static __device__ __forceinline__ v8f wmma_bf16(v16bf a, v16bf b, v8f c) {
    return __builtin_amdgcn_wmma_f32_16x16x32_bf16(
        false, a, false, b, (short)0, c, false, false);
}

// Butterfly reductions inside each 16-lane row via v_permlane16 (VALU, no LDS)
static __device__ __forceinline__ float permx16(float v, u32 s0, u32 s1) {
#if HAVE_PERMLANE16
    u32 u = __float_as_uint(v);
    u32 r = __builtin_amdgcn_permlane16(u, u, s0, s1, false, false);
    return __uint_as_float(r);
#else
    (void)s0; (void)s1;
    return v;
#endif
}
static __device__ __forceinline__ float rowmax16(float v) {
#if HAVE_PERMLANE16
    v = fmaxf(v, permx16(v, 0x67452301u, 0xEFCDAB89u)); // xor 1
    v = fmaxf(v, permx16(v, 0x54761032u, 0xDCFE98BAu)); // xor 2
    v = fmaxf(v, permx16(v, 0x32107654u, 0xBA98FEDCu)); // xor 4
    v = fmaxf(v, permx16(v, 0xFEDCBA98u, 0x76543210u)); // xor 8
#else
#pragma unroll
    for (int off = 1; off < 16; off <<= 1) v = fmaxf(v, __shfl_xor(v, off, 32));
#endif
    return v;
}
static __device__ __forceinline__ float rowsum16(float v) {
#if HAVE_PERMLANE16
    v += permx16(v, 0x67452301u, 0xEFCDAB89u);
    v += permx16(v, 0x54761032u, 0xDCFE98BAu);
    v += permx16(v, 0x32107654u, 0xBA98FEDCu);
    v += permx16(v, 0xFEDCBA98u, 0x76543210u);
#else
#pragma unroll
    for (int off = 1; off < 16; off <<= 1) v += __shfl_xor(v, off, 32);
#endif
    return v;
}

// ---------------------------------------------------------------------------
// Tensor Data Mover: 2D tile (bf16) -> padded LDS.  One issue per wave.
//   tile_d0 elements contiguous (dim0), tile_d1 rows, row stride in elements.
//   pad_interval code: pad every 2^(c+1) DWORDs ; pad_amount code: (c+1) DWORDs
// ---------------------------------------------------------------------------
static __device__ __forceinline__ void tdm_load_tile_2d(
    u32 lds_addr, u64 gaddr, u32 tile_d0, u32 tile_d1, u64 stride_elems,
    u32 pad_interval, u32 pad_amount) {
#if HAVE_TDM
    union { u32 w[4]; u32x4 v; } g0;
    g0.w[0] = 1u;                                   // count=1 (valid), load
    g0.w[1] = lds_addr;                             // LDS byte offset
    g0.w[2] = (u32)(gaddr & 0xFFFFFFFFull);         // global addr lo
    g0.w[3] = (u32)((gaddr >> 32) & 0x1FFFFFFull)   // global addr hi (57b)
            | (2u << 30);                           // type = 2 (image/D#)
    union { u64 q[4]; i32x8 v; } g1;
    g1.q[0] = (1ull << 16)                          // data_size = 2 bytes
            | (1ull << 20)                          // pad_enable
            | ((u64)pad_interval << 22)
            | ((u64)pad_amount << 25)
            | ((u64)(tile_d0 & 0xFFFFu) << 48);     // tensor_dim0[15:0]
    g1.q[1] = ((u64)tile_d0 >> 16)                  // tensor_dim0[31:16]
            | ((u64)tile_d1 << 16)                  // tensor_dim1
            | ((u64)tile_d0 << 48);                 // tile_dim0
    g1.q[2] = ((u64)tile_d1)                        // tile_dim1 (tile_dim2=0)
            | ((stride_elems & 0xFFFFFFFFull) << 32); // dim0 stride lo32
    g1.q[3] = (stride_elems >> 32) & 0xFFFFull;     // dim0 stride hi16
    i32x4 z4 = {0, 0, 0, 0};
#if TDM_SIX_ARGS
    i32x8 z8 = {0, 0, 0, 0, 0, 0, 0, 0};
    __builtin_amdgcn_tensor_load_to_lds(g0.v, g1.v, z4, z4, z8, 0);
#else
    __builtin_amdgcn_tensor_load_to_lds(g0.v, g1.v, z4, z4, 0);
#endif
#else
    (void)lds_addr; (void)gaddr; (void)tile_d0; (void)tile_d1;
    (void)stride_elems; (void)pad_interval; (void)pad_amount;
#endif
}

// ---------------------------------------------------------------------------
// Kernel 1: fp32 -> bf16 conversion (vectorized)
// ---------------------------------------------------------------------------
__global__ __launch_bounds__(256) void cvt_f32_bf16(
    const float* __restrict__ in, __bf16* __restrict__ out, int n4) {
    int i = blockIdx.x * blockDim.x + threadIdx.x;
    if (i < n4) {
        float4 f = ((const float4*)in)[i];
        union { __bf16 b[4]; int2 v; } u;
        u.b[0] = (__bf16)f.x; u.b[1] = (__bf16)f.y;
        u.b[2] = (__bf16)f.z; u.b[3] = (__bf16)f.w;
        ((int2*)out)[i] = u.v;
    }
}

// ---------------------------------------------------------------------------
// Kernel 2: tiled bf16 WMMA GEMM, C[M,N] = A[M,K] @ B[K,N] (+ bias)
//   256 threads = 8 waves, 128x128 C tile, BK=32, wave grid 4x2 (32x64/wave).
//   A tile staged by the Tensor Data Mover, double buffered so the DMA of
//   tile k+1 overlaps the WMMAs of tile k.  B tile transposed manually.
// ---------------------------------------------------------------------------
template <bool BF16_OUT>
__global__ __launch_bounds__(256) void gemm_wmma(
    const __bf16* __restrict__ A, const __bf16* __restrict__ B,
    void* __restrict__ Cout, const float* __restrict__ bias,
    int M, int N, int K) {
#if HAVE_TDM
    __shared__ __bf16 As[2][128][40];   // [buf][M][K] padded (32+8 elems/row)
#else
    __shared__ __bf16 As[1][128][40];
#endif
    __shared__ __bf16 BsT[128][40];     // [N][K] transposed, padded

    const int tid  = threadIdx.x;
    const int lane = tid & 31;
    const int wid  = tid >> 5;
    const int half = (lane >> 4) & 1;
    const int lm   = lane & 15;
    const int wm   = wid >> 1;   // 0..3
    const int wn   = wid & 1;    // 0..1
    const int rowBase = blockIdx.y * 128;
    const int colBase = blockIdx.x * 128;

    v8f acc[2][4];
#pragma unroll
    for (int mt = 0; mt < 2; ++mt)
#pragma unroll
        for (int nt = 0; nt < 4; ++nt) acc[mt][nt] = v8f_zero();

#if HAVE_TDM
    if (wid == 0) {
        tdm_load_tile_2d((u32)(uintptr_t)&As[0][0][0],
                         (u64)(uintptr_t)(A + (size_t)rowBase * K),
                         /*d0=*/32, /*d1=*/128, /*stride=*/(u64)K,
                         /*pad_int 16DW=*/3, /*pad_amt 4DW=*/3);
    }
#endif

    for (int kb = 0; kb < K; kb += 32) {
#if HAVE_TDM
        const int cur = (kb >> 5) & 1;
        if (wid == 0 && kb + 32 < K) {
            tdm_load_tile_2d((u32)(uintptr_t)&As[cur ^ 1][0][0],
                             (u64)(uintptr_t)(A + (size_t)rowBase * K + kb + 32),
                             32, 128, (u64)K, 3, 3);
        }
#else
        const int cur = 0;
        // ---- stage A tile (128x32): 512 x 16B chunks / 256 threads ----
#pragma unroll
        for (int it = 0; it < 2; ++it) {
            int c  = tid + 256 * it;
            int r  = c >> 2;
            int cc = (c & 3) * 8;
            int4 d = *(const int4*)(A + (size_t)(rowBase + r) * K + kb + cc);
            *(int4*)&As[0][r][cc] = d;
        }
#endif
        // ---- stage B tile (32x128), transposed into BsT[N][K] ----
#pragma unroll
        for (int it = 0; it < 2; ++it) {
            int c  = tid + 256 * it;
            int kr = c >> 4;
            int cc = (c & 15) * 8;
            union { int4 i; __bf16 b[8]; } u;
            u.i = *(const int4*)(B + (size_t)(kb + kr) * N + colBase + cc);
#pragma unroll
            for (int e = 0; e < 8; ++e) BsT[cc + e][kr] = u.b[e];
        }
        if (kb + 32 < K)
            __builtin_prefetch(B + (size_t)(kb + 32 + (tid >> 3)) * N + colBase, 0, 3);
#if HAVE_TDM
        if (wid == 0) {
            if (kb + 32 < K) WAIT_TENSORCNT(1);  // current tile done
            else             WAIT_TENSORCNT(0);
        }
#endif
        __syncthreads();

        const __bf16(*Ac)[40] = As[cur];
        v16bf af[2], bfr[4];
#pragma unroll
        for (int mt = 0; mt < 2; ++mt)
            af[mt] = frag_a(&Ac[wm * 32 + mt * 16 + lm][0], half);
#pragma unroll
        for (int nt = 0; nt < 4; ++nt)
            bfr[nt] = frag_b(&BsT[wn * 64 + nt * 16 + lm][0], half);

#pragma unroll
        for (int mt = 0; mt < 2; ++mt)
#pragma unroll
            for (int nt = 0; nt < 4; ++nt)
                acc[mt][nt] = wmma_bf16(af[mt], bfr[nt], acc[mt][nt]);
        __syncthreads();
    }

    // ---- epilogue: C layout lane=col(L%16), vgpr r -> row r + 8*(L/16) ----
#pragma unroll
    for (int mt = 0; mt < 2; ++mt)
#pragma unroll
        for (int nt = 0; nt < 4; ++nt)
#pragma unroll
            for (int r = 0; r < 8; ++r) {
                int row = rowBase + wm * 32 + mt * 16 + r + 8 * half;
                int col = colBase + wn * 64 + nt * 16 + lm;
                float v = acc[mt][nt][r];
                if constexpr (BF16_OUT) {
                    ((__bf16*)Cout)[(size_t)row * N + col] = (__bf16)v;
                } else {
                    ((float*)Cout)[(size_t)row * N + col] = v + bias[col];
                }
            }
}

// ---------------------------------------------------------------------------
// Kernel 3: causal flash-attention core (per (b,h), 64 q-rows per block)
//   q:(B*N, H*Dh)  kv:(B*N, 2*H*Dh)  bf16, from the GEMMs above.
//   128 threads = 4 waves; wave w owns rows q0 = qblk + 16w.
//   K tile staged via TDM (wave 0) while all waves transpose-stage V.
// ---------------------------------------------------------------------------
__global__ __launch_bounds__(128) void attn_causal(
    const __bf16* __restrict__ q, const __bf16* __restrict__ kv,
    __bf16* __restrict__ attn) {
    __shared__ __bf16 Kls[32][72];       // [key][dh], padded (64+8)
    __shared__ __bf16 VlsT[64][40];      // [dh][key], padded (32+8)
    __shared__ __bf16 Pls[4][16][40];    // per-wave P bounce buffer

    const int tid  = threadIdx.x;
    const int lane = tid & 31;
    const int wid  = tid >> 5;
    const int half = (lane >> 4) & 1;
    const int lm   = lane & 15;

    const int bh   = blockIdx.y;
    const int b    = bh >> 4;
    const int h    = bh & 15;
    const int qblk = blockIdx.x * 64;
    const int q0   = qblk + wid * 16;

    const float scale = 0.125f; // 1/sqrt(64)

    // Q fragments straight from global (A layout): lane row = q0 + lm
    const __bf16* qrow = q + ((size_t)(b * NN + q0 + lm)) * INNER + h * DHD;
    v16bf qf[2];
#pragma unroll
    for (int ks = 0; ks < 2; ++ks) qf[ks] = frag_a(qrow + ks * 32, half);

    float m[8], l[8];
    v8f oacc[4];
#pragma unroll
    for (int r = 0; r < 8; ++r) { m[r] = -3.0e38f; l[r] = 0.0f; }
#pragma unroll
    for (int t = 0; t < 4; ++t) oacc[t] = v8f_zero();

    const int jb_end = qblk + 64;
    for (int jb = 0; jb < jb_end; jb += 32) {
        const size_t kvrow0 = ((size_t)(b * NN + jb)) * (2 * INNER) + h * DHD;
#if HAVE_TDM
        if (wid == 0) {
            // K tile (32 keys x 64 dh) -> Kls, padded rows (64+8 elems)
            tdm_load_tile_2d((u32)(uintptr_t)&Kls[0][0],
                             (u64)(uintptr_t)(kv + kvrow0),
                             /*d0=*/64, /*d1=*/32, /*stride=*/(u64)(2 * INNER),
                             /*pad_int 32DW=*/4, /*pad_amt 4DW=*/3);
        }
#endif
        // ---- stage V^T (64x32) cooperatively (and K if no TDM) ----
#pragma unroll
        for (int it = 0; it < 2; ++it) {
            int c  = tid + 128 * it;
            int kr = c >> 3;          // key row 0..31
            int dc = (c & 7) * 8;     // dh col 0..56
#if !HAVE_TDM
            int4 kd = *(const int4*)(kv + kvrow0 + (size_t)kr * (2 * INNER) + dc);
            *(int4*)&Kls[kr][dc] = kd;
#endif
            union { int4 i; __bf16 bv[8]; } u;
            u.i = *(const int4*)(kv + kvrow0 + INNER + (size_t)kr * (2 * INNER) + dc);
#pragma unroll
            for (int e = 0; e < 8; ++e) VlsT[dc + e][kr] = u.bv[e];
        }
#if HAVE_TDM
        if (wid == 0) WAIT_TENSORCNT(0);
#endif
        __syncthreads();

        const bool active = (jb <= q0 + 15);  // wave-uniform
        if (active) {
            // ---- S = Q @ K^T (16 rows x 32 keys = two 16x16 tiles) ----
            v8f sa[2];
#pragma unroll
            for (int ct = 0; ct < 2; ++ct) {
                v8f s = v8f_zero();
#pragma unroll
                for (int ks = 0; ks < 2; ++ks) {
                    v16bf kf = frag_b(&Kls[ct * 16 + lm][ks * 32], half);
                    s = wmma_bf16(qf[ks], kf, s);
                }
                sa[ct] = s;
            }
            // ---- scale (+ causal mask only on diagonal blocks) ----
#pragma unroll
            for (int r = 0; r < 8; ++r) {
                sa[0][r] *= scale;
                sa[1][r] *= scale;
            }
            if (jb + 31 > q0) {        // wave-uniform: block touches diagonal
#pragma unroll
                for (int r = 0; r < 8; ++r) {
                    int ig = q0 + r + 8 * half;
                    if (jb + lm > ig)      sa[0][r] = -3.0e38f;
                    if (jb + 16 + lm > ig) sa[1][r] = -3.0e38f;
                }
            }
            // ---- online softmax ----
            float fac[8];
#pragma unroll
            for (int r = 0; r < 8; ++r) {
                float mx = rowmax16(fmaxf(sa[0][r], sa[1][r]));
                float mn = fmaxf(m[r], mx);
                fac[r]   = __expf(m[r] - mn);
                m[r]     = mn;
                float p0 = __expf(sa[0][r] - mn);
                float p1 = __expf(sa[1][r] - mn);
                sa[0][r] = p0; sa[1][r] = p1;
                l[r] = l[r] * fac[r] + rowsum16(p0 + p1);
            }
            // ---- bounce P through LDS: C layout -> A layout, as bf16 ----
#pragma unroll
            for (int ct = 0; ct < 2; ++ct)
#pragma unroll
                for (int r = 0; r < 8; ++r)
                    Pls[wid][r + 8 * half][ct * 16 + lm] = (__bf16)sa[ct][r];
            asm volatile("s_wait_dscnt 0" ::: "memory");  // wave-local RAW
            v16bf pf = frag_a(&Pls[wid][lm][0], half);
            // ---- O += P @ V (four 16x16 dh tiles, K = 32 keys) ----
#pragma unroll
            for (int t = 0; t < 4; ++t) {
#pragma unroll
                for (int r = 0; r < 8; ++r) oacc[t][r] *= fac[r];
                v16bf vf = frag_b(&VlsT[t * 16 + lm][0], half);
                oacc[t] = wmma_bf16(pf, vf, oacc[t]);
            }
        }
        __syncthreads();
    }

    // ---- epilogue: out = acc / (l + eps), store bf16 (B*N, H*Dh) ----
#pragma unroll
    for (int t = 0; t < 4; ++t)
#pragma unroll
        for (int r = 0; r < 8; ++r) {
            int ig = q0 + r + 8 * half;
            int d  = t * 16 + lm;
            float v = oacc[t][r] / (l[r] + 1e-8f);
            attn[((size_t)(b * NN + ig)) * INNER + h * DHD + d] = (__bf16)v;
        }
}

// ---------------------------------------------------------------------------
// Host launcher
// ---------------------------------------------------------------------------
extern "C" void kernel_launch(void* const* d_in, const int* in_sizes, int n_in,
                              void* d_out, int out_size, void* d_ws, size_t ws_size,
                              hipStream_t stream) {
    const float* x   = (const float*)d_in[0];
    const float* Wq  = (const float*)d_in[1];
    const float* Wkv = (const float*)d_in[2];
    const float* Wo  = (const float*)d_in[3];
    const float* bo  = (const float*)d_in[4];
    float* out = (float*)d_out;

    char* ws = (char*)d_ws;
    size_t off = 0;
    __bf16* xbf   = (__bf16*)(ws + off); off += (size_t)MROWS * DD * 2;        // 8 MB
    __bf16* wqbf  = (__bf16*)(ws + off); off += (size_t)DD * INNER * 2;        // 2 MB
    __bf16* wkvbf = (__bf16*)(ws + off); off += (size_t)DD * 2 * INNER * 2;    // 4 MB
    __bf16* wobf  = (__bf16*)(ws + off); off += (size_t)INNER * DD * 2;        // 2 MB
    __bf16* qbf   = (__bf16*)(ws + off); off += (size_t)MROWS * INNER * 2;     // 8 MB
    __bf16* kvbf  = (__bf16*)(ws + off); off += (size_t)MROWS * 2 * INNER * 2; // 16 MB
    __bf16* abf   = (__bf16*)(ws + off); off += (size_t)MROWS * INNER * 2;     // 8 MB

    // 1) fp32 -> bf16 conversions
    {
        int n4;
        n4 = MROWS * DD / 4;
        cvt_f32_bf16<<<(n4 + 255) / 256, 256, 0, stream>>>(x, xbf, n4);
        n4 = DD * INNER / 4;
        cvt_f32_bf16<<<(n4 + 255) / 256, 256, 0, stream>>>(Wq, wqbf, n4);
        n4 = DD * 2 * INNER / 4;
        cvt_f32_bf16<<<(n4 + 255) / 256, 256, 0, stream>>>(Wkv, wkvbf, n4);
        n4 = INNER * DD / 4;
        cvt_f32_bf16<<<(n4 + 255) / 256, 256, 0, stream>>>(Wo, wobf, n4);
    }

    // 2) Q = x @ Wq      (4096 x 1024 x 1024)
    gemm_wmma<true><<<dim3(INNER / 128, MROWS / 128), 256, 0, stream>>>(
        xbf, wqbf, (void*)qbf, nullptr, MROWS, INNER, DD);

    // 3) KV = x @ Wkv    (4096 x 2048 x 1024)
    gemm_wmma<true><<<dim3(2 * INNER / 128, MROWS / 128), 256, 0, stream>>>(
        xbf, wkvbf, (void*)kvbf, nullptr, MROWS, 2 * INNER, DD);

    // 4) causal flash attention
    attn_causal<<<dim3(NN / 64, BB * HH), 128, 0, stream>>>(qbf, kvbf, abf);

    // 5) out = attn @ Wo + bo  (4096 x 1024 x 1024), fp32
    gemm_wmma<false><<<dim3(DD / 128, MROWS / 128), 256, 0, stream>>>(
        abf, wobf, (void*)out, bo, MROWS, DD, INNER);

    (void)in_sizes; (void)n_in; (void)out_size; (void)ws_size;
}
